// Expert_17051020165440
// MI455X (gfx1250) — compile-verified, matrix operand order checked by reading, not run on hardware
//
#include <hip/hip_runtime.h>
#include <hip/hip_bf16.h>
#include <stdint.h>

#define HIDDEN 4096
#define INTER  14336
#define NSEL   4096
#define KC     32
#define TILE   128
#define LSTRIDE 40   // KC + 8 bf16 pad: avoids LDS bank conflicts, keeps 16B alignment

typedef __attribute__((ext_vector_type(16))) __bf16 v16bf;
typedef __attribute__((ext_vector_type(8)))  __bf16 v8bf;
typedef __attribute__((ext_vector_type(8)))  float  v8f;

// round-to-nearest-even fp32 -> bf16 (scalar, epilogue use)
__device__ __forceinline__ __bf16 f2bf(float f) {
    uint32_t u = __float_as_uint(f);
    u += 0x7FFFu + ((u >> 16) & 1u);
    return __builtin_bit_cast(__bf16, (uint16_t)(u >> 16));
}

// pack two fp32 -> one dword of 2x bf16 (round-to-nearest via +0x8000, then
// one v_perm_b32 packs the two high halves: 3 VALU ops per pair)
__device__ __forceinline__ uint32_t pk_bf16(float lo, float hi) {
    uint32_t ul = __float_as_uint(lo) + 0x8000u;
    uint32_t uh = __float_as_uint(hi) + 0x8000u;
    // bytes 0-3 select from ul, 4-7 from uh: take ul[3:2] then uh[3:2]
    return __builtin_amdgcn_perm(uh, ul, 0x07060302u);
}

// convert 8 fp32 (two float4) -> 8 bf16, one b128 LDS/global store
__device__ __forceinline__ void cvt_store8(float4 a, float4 b, __bf16* dst) {
    uint4 w;
    w.x = pk_bf16(a.x, a.y);
    w.y = pk_bf16(a.z, a.w);
    w.z = pk_bf16(b.x, b.y);
    w.w = pk_bf16(b.z, b.w);
    *(uint4*)dst = w;
}

// Load one 16x32 bf16 A/B fragment from an LDS tile (base = &lds[row0*LSTRIDE]).
// ISA layout: lanes 0-15 hold K=0..7 and 16..23; lanes 16-31 hold K=8..15 and 24..31.
__device__ __forceinline__ v16bf ld_frag(const __bf16* base, int m, int half) {
    const __bf16* q = base + m * LSTRIDE;
    v16bf f;
    ((v8bf*)&f)[0] = *(const v8bf*)(q + half * 8);
    ((v8bf*)&f)[1] = *(const v8bf*)(q + 16 + half * 8);
    return f;
}

__device__ __forceinline__ v8f wmma_bf16(v16bf a, v16bf b, v8f c) {
    return __builtin_amdgcn_wmma_f32_16x16x32_bf16(false, a, false, b, (short)0, c,
                                                   false, false);
}

// fast SiLU: g * sigmoid(g) with v_rcp_f32 (no IEEE div expansion)
__device__ __forceinline__ float fast_silu(float g) {
    return g * __builtin_amdgcn_rcpf(1.f + __expf(-g));
}

// ---------------------------------------------------------------- gather+cast
__global__ __launch_bounds__(256) void k_gather(const float* __restrict__ x,
                                               const long long* __restrict__ topx,
                                               __bf16* __restrict__ xsel) {
    int idx  = blockIdx.x * 256 + threadIdx.x;      // one thread = 8 elements
    int row  = idx >> 9;                            // HIDDEN/8 = 512 chunks per row
    int col  = (idx & 511) * 8;
    long long src = topx[row];
    const float4* p = (const float4*)(x + (size_t)src * HIDDEN + col);
    cvt_store8(p[0], p[1], xsel + (size_t)row * HIDDEN + col);
}

// ------------------------------------------------- fused gate/up GEMM + SwiGLU
// C_g = Xsel * Wg^T, C_u = Xsel * Wu^T  (NT GEMM over K=HIDDEN)
__global__ __launch_bounds__(512, 1) void k_gateup(const __bf16* __restrict__ xsel,
                                                   const float* __restrict__ Wg,
                                                   const float* __restrict__ Wu,
                                                   __bf16* __restrict__ hws) {
    __shared__ __bf16 lA[TILE * LSTRIDE];
    __shared__ __bf16 lG[TILE * LSTRIDE];
    __shared__ __bf16 lU[TILE * LSTRIDE];

    const int tid  = threadIdx.x;
    const int lane = tid & 31;
    const int wid  = tid >> 5;            // 0..15 waves
    const int m    = lane & 15;
    const int half = lane >> 4;
    const int wn   = (wid >> 2) * 32;     // wave row offset in 128x128 block tile
    const int wi   = (wid & 3) * 32;      // wave col offset

    const int nbase = blockIdx.x * TILE;  // token rows
    const int ibase = blockIdx.y * TILE;  // inter cols

    const int srow = (tid * 8) >> 5;      // staging: 8 elems/thread, 128x32 tile
    const int scol = (tid * 8) & 31;

    const __bf16* gA = xsel + (size_t)(nbase + srow) * HIDDEN + scol;
    const float*  gG = Wg   + (size_t)(ibase + srow) * HIDDEN + scol;
    const float*  gU = Wu   + (size_t)(ibase + srow) * HIDDEN + scol;

    v8f accg[2][2], accu[2][2];
#pragma unroll
    for (int r = 0; r < 2; ++r)
#pragma unroll
        for (int c = 0; c < 2; ++c) {
            accg[r][c] = (v8f){0.f, 0.f, 0.f, 0.f, 0.f, 0.f, 0.f, 0.f};
            accu[r][c] = (v8f){0.f, 0.f, 0.f, 0.f, 0.f, 0.f, 0.f, 0.f};
        }

    // software pipeline: prefetch K-panel 0 into registers
    uint4  pa  = *(const uint4*)gA;
    float4 pg0 = ((const float4*)gG)[0], pg1 = ((const float4*)gG)[1];
    float4 pu0 = ((const float4*)gU)[0], pu1 = ((const float4*)gU)[1];

    for (int k0 = 0; k0 < HIDDEN; k0 += KC) {
        // drain staged registers into LDS
        *(uint4*)(lA + srow * LSTRIDE + scol) = pa;
        cvt_store8(pg0, pg1, lG + srow * LSTRIDE + scol);
        cvt_store8(pu0, pu1, lU + srow * LSTRIDE + scol);
        __syncthreads();

        // prefetch next K-panel (latency hidden under WMMAs below)
        if (k0 + KC < HIDDEN) {
            pa  = *(const uint4*)(gA + k0 + KC);
            pg0 = ((const float4*)(gG + k0 + KC))[0];
            pg1 = ((const float4*)(gG + k0 + KC))[1];
            pu0 = ((const float4*)(gU + k0 + KC))[0];
            pu1 = ((const float4*)(gU + k0 + KC))[1];
        }

        v16bf a[2], bg[2], bu[2];
        a[0]  = ld_frag(lA + (wn + 0)  * LSTRIDE, m, half);
        a[1]  = ld_frag(lA + (wn + 16) * LSTRIDE, m, half);
        bg[0] = ld_frag(lG + (wi + 0)  * LSTRIDE, m, half);
        bg[1] = ld_frag(lG + (wi + 16) * LSTRIDE, m, half);
        bu[0] = ld_frag(lU + (wi + 0)  * LSTRIDE, m, half);
        bu[1] = ld_frag(lU + (wi + 16) * LSTRIDE, m, half);

#pragma unroll
        for (int r = 0; r < 2; ++r)
#pragma unroll
            for (int c = 0; c < 2; ++c) {
                accg[r][c] = wmma_bf16(a[r], bg[c], accg[r][c]);
                accu[r][c] = wmma_bf16(a[r], bu[c], accu[r][c]);
            }
        __syncthreads();
    }

    // epilogue: h = silu(g) * u -> bf16 workspace
#pragma unroll
    for (int r = 0; r < 2; ++r)
#pragma unroll
        for (int c = 0; c < 2; ++c)
#pragma unroll
            for (int v = 0; v < 8; ++v) {
                int row = nbase + wn + r * 16 + v + 8 * half;
                int col = ibase + wi + c * 16 + m;
                float h = fast_silu(accg[r][c][v]) * accu[r][c][v];
                hws[(size_t)row * INTER + col] = f2bf(h);
            }
}

// ------------------------------------------------------- down GEMM + weighting
// out = (H * Wd^T) * weight[n],  K = INTER
__global__ __launch_bounds__(512, 1) void k_down(const __bf16* __restrict__ hws,
                                                 const float* __restrict__ Wd,
                                                 const float* __restrict__ weight,
                                                 float* __restrict__ out) {
    __shared__ __bf16 lA[TILE * LSTRIDE];
    __shared__ __bf16 lB[TILE * LSTRIDE];

    const int tid  = threadIdx.x;
    const int lane = tid & 31;
    const int wid  = tid >> 5;
    const int m    = lane & 15;
    const int half = lane >> 4;
    const int wn   = (wid >> 2) * 32;
    const int wh   = (wid & 3) * 32;

    const int nbase = blockIdx.x * TILE;
    const int hbase = blockIdx.y * TILE;

    const int srow = (tid * 8) >> 5;
    const int scol = (tid * 8) & 31;

    const __bf16* gA = hws + (size_t)(nbase + srow) * INTER + scol;
    const float*  gB = Wd  + (size_t)(hbase + srow) * INTER + scol;

    v8f acc[2][2];
#pragma unroll
    for (int r = 0; r < 2; ++r)
#pragma unroll
        for (int c = 0; c < 2; ++c)
            acc[r][c] = (v8f){0.f, 0.f, 0.f, 0.f, 0.f, 0.f, 0.f, 0.f};

    uint4  pa  = *(const uint4*)gA;
    float4 pb0 = ((const float4*)gB)[0], pb1 = ((const float4*)gB)[1];

    for (int k0 = 0; k0 < INTER; k0 += KC) {
        *(uint4*)(lA + srow * LSTRIDE + scol) = pa;
        cvt_store8(pb0, pb1, lB + srow * LSTRIDE + scol);
        __syncthreads();

        if (k0 + KC < INTER) {
            pa  = *(const uint4*)(gA + k0 + KC);
            pb0 = ((const float4*)(gB + k0 + KC))[0];
            pb1 = ((const float4*)(gB + k0 + KC))[1];
        }

        v16bf a[2], b[2];
        a[0] = ld_frag(lA + (wn + 0)  * LSTRIDE, m, half);
        a[1] = ld_frag(lA + (wn + 16) * LSTRIDE, m, half);
        b[0] = ld_frag(lB + (wh + 0)  * LSTRIDE, m, half);
        b[1] = ld_frag(lB + (wh + 16) * LSTRIDE, m, half);

#pragma unroll
        for (int r = 0; r < 2; ++r)
#pragma unroll
            for (int c = 0; c < 2; ++c)
                acc[r][c] = wmma_bf16(a[r], b[c], acc[r][c]);
        __syncthreads();
    }

#pragma unroll
    for (int r = 0; r < 2; ++r)
#pragma unroll
        for (int c = 0; c < 2; ++c)
#pragma unroll
            for (int v = 0; v < 8; ++v) {
                int row = nbase + wn + r * 16 + v + 8 * half;
                int col = hbase + wh + c * 16 + m;
                out[(size_t)row * HIDDEN + col] = acc[r][c][v] * weight[row];
            }
}

extern "C" void kernel_launch(void* const* d_in, const int* in_sizes, int n_in,
                              void* d_out, int out_size, void* d_ws, size_t ws_size,
                              hipStream_t stream) {
    (void)in_sizes; (void)n_in; (void)out_size; (void)ws_size;
    const float*     x      = (const float*)d_in[0];
    const float*     Wg     = (const float*)d_in[1];
    const float*     Wu     = (const float*)d_in[2];
    const float*     Wd     = (const float*)d_in[3];
    const long long* topx   = (const long long*)d_in[4];
    const float*     weight = (const float*)d_in[5];
    float*           out    = (float*)d_out;

    __bf16* xsel = (__bf16*)d_ws;                                      // 33.5 MB
    __bf16* hws  = (__bf16*)((char*)d_ws + (size_t)NSEL * HIDDEN * 2); // 117.4 MB

    // 1) gather selected tokens, cast to bf16
    k_gather<<<(NSEL * HIDDEN) / (256 * 8), 256, 0, stream>>>(x, topx, xsel);

    // 2) fused gate/up GEMM + SwiGLU (x-major block order keeps weight panel in L2)
    dim3 g2(NSEL / TILE, INTER / TILE);
    k_gateup<<<g2, 512, 0, stream>>>(xsel, Wg, Wu, hws);

    // 3) down projection + per-token routing weight
    dim3 g3(NSEL / TILE, HIDDEN / TILE);
    k_down<<<g3, 512, 0, stream>>>(hws, Wd, weight, out);
}
